// GraphPDE_75462575390928
// MI455X (gfx1250) — matrix operand order repaired and checked by reading, not compiled, not run on hardware
//
#include <hip/hip_runtime.h>
#include <hip/hip_bf16.h>

typedef __attribute__((ext_vector_type(16))) _Float16 v16h;
typedef __attribute__((ext_vector_type(8)))  _Float16 v8h;
typedef __attribute__((ext_vector_type(8)))  float    v8f;

#define NWAVES 8
#define HS 72   // LDS staging row stride in halves: 144B = 16B-aligned, 36 dwords per row

// ---------------------------------------------------------------------------
// D = A(16x32 f16) * B(32x16 f16) + C(16x16 f32)
// ---------------------------------------------------------------------------
__device__ __forceinline__ v8f wmma_f16(v16h a, v16h b, v8f c) {
    return __builtin_amdgcn_wmma_f32_16x16x32_f16(false, a, false, b, (short)0, c,
                                                  false, false);
}

// Build a B-fragment (32x16 f16) for v_wmma_f32_16x16x32_f16 from a row-major
// f32 weight matrix W[k][64].  Lane = hi*16 + n; element e <-> K = kbase+8*hi+e (e<8), kbase+16+8*hi+(e-8) (e>=8),
// column = ncol + n.  Rows >= krows are zero padding.
__device__ __forceinline__ v16h load_bfrag(const float* __restrict__ W, int krows,
                                           int kbase, int ncol, int lane) {
    const int hi = (lane >> 4) & 1;
    const int n  = lane & 15;
    v16h b;
#pragma unroll
    for (int e = 0; e < 16; ++e) {
        const int k = kbase + (e < 8 ? hi * 8 + e : 16 + hi * 8 + (e - 8));
        b[e] = (k < krows) ? (_Float16)W[k * 64 + ncol + n] : (_Float16)0.0f;
    }
    return b;
}

// Shared 3-layer MLP tail:  A0(16x32) -> 64 (relu) -> 64 (relu) -> 1
// Returns per-row scalar output (valid in lanes 0..15, row = lane).
// hrow is a wave-private LDS tile of 16*HS halves.
__device__ __forceinline__ float mlp3(v16h a0,
                                      const v16h B0[4], const v16h B1[2][4],
                                      const float* __restrict__ b0,
                                      const float* __restrict__ b1,
                                      const float* __restrict__ w2s, float b2,
                                      _Float16* hrow, int lane) {
    const int hi = (lane >> 4) & 1;
    const int n  = lane & 15;
    v8f cz;
#pragma unroll
    for (int i = 0; i < 8; ++i) cz[i] = 0.0f;

    // ---- layer 0: 16x32 @ 32x64 ----
    v8f c[4];
#pragma unroll
    for (int nt = 0; nt < 4; ++nt) c[nt] = wmma_f16(a0, B0[nt], cz);

    // bias + relu, stage H1 to LDS as f16 [m][k]  (C/D layout: elem r -> M=r+8*hi, N=n)
#pragma unroll
    for (int nt = 0; nt < 4; ++nt) {
        const float bn = b0[nt * 16 + n];
#pragma unroll
        for (int r = 0; r < 8; ++r) {
            float v = c[nt][r] + bn;
            v = v > 0.0f ? v : 0.0f;
            hrow[(r + 8 * hi) * HS + nt * 16 + n] = (_Float16)v;
        }
    }

    // gather layer-1 A fragments (two K-chunks of 32) via 16B ds loads
    v16h a1[2];
#pragma unroll
    for (int kk = 0; kk < 2; ++kk) {
        const v8h lo = *(const v8h*)&hrow[n * HS + kk * 32 + 8 * hi];
        const v8h hh = *(const v8h*)&hrow[n * HS + kk * 32 + 16 + 8 * hi];
#pragma unroll
        for (int i = 0; i < 8; ++i) { a1[kk][i] = lo[i]; a1[kk][i + 8] = hh[i]; }
    }

    // ---- layer 1: 16x64 @ 64x64 ----
    v8f d[4];
#pragma unroll
    for (int nt = 0; nt < 4; ++nt) {
        d[nt] = wmma_f16(a1[0], B1[0][nt], cz);
        d[nt] = wmma_f16(a1[1], B1[1][nt], d[nt]);
    }

    // bias + relu, stage H2 (reuse wave-private tile; DS ops are in-order per wave)
#pragma unroll
    for (int nt = 0; nt < 4; ++nt) {
        const float bn = b1[nt * 16 + n];
#pragma unroll
        for (int r = 0; r < 8; ++r) {
            float v = d[nt][r] + bn;
            v = v > 0.0f ? v : 0.0f;
            hrow[(r + 8 * hi) * HS + nt * 16 + n] = (_Float16)v;
        }
    }

    // ---- layer 2: 64 -> 1 dot, split across half-waves ----
    float p = 0.0f;
#pragma unroll
    for (int k = 0; k < 32; ++k)
        p += (float)hrow[n * HS + hi * 32 + k] * w2s[hi * 32 + k];
    p += __shfl_xor(p, 16);
    return p + b2;
}

// ---------------------------------------------------------------------------
__global__ void __launch_bounds__(256)
zero_kernel(float* __restrict__ p, int n) {
    int i = blockIdx.x * blockDim.x + threadIdx.x;
    if (i < n) p[i] = 0.0f;
}

// phi over edges + atomic mean-aggregation onto dst nodes
__global__ void __launch_bounds__(256)
phi_edge_kernel(const float* __restrict__ x, const float* __restrict__ pos,
                const int* __restrict__ ei,
                const float* __restrict__ W0, const float* __restrict__ b0,
                const float* __restrict__ W1, const float* __restrict__ b1,
                const float* __restrict__ W2, const float* __restrict__ b2,
                float* __restrict__ agg, float* __restrict__ cnt,
                int nE, int nTiles) {
    __shared__ _Float16 hbuf[NWAVES][16 * HS];
    __shared__ float w2s[64];

    const int tid  = threadIdx.x;
    const int lane = tid & 31;
    const int wave = tid >> 5;
    const int hi   = (lane >> 4) & 1;
    const int n    = lane & 15;

    if (tid < 64) w2s[tid] = W2[tid];
    __syncthreads();

    // preload weight fragments (f32 -> f16), reused across all tiles
    v16h B0f[4], B1f[2][4];
#pragma unroll
    for (int nt = 0; nt < 4; ++nt) B0f[nt] = load_bfrag(W0, 18, 0, nt * 16, lane);
#pragma unroll
    for (int kk = 0; kk < 2; ++kk)
#pragma unroll
        for (int nt = 0; nt < 4; ++nt)
            B1f[kk][nt] = load_bfrag(W1, 64, kk * 32, nt * 16, lane);
    const float bias2 = b2[0];
    _Float16* hrow = hbuf[wave];

    const int wg = blockIdx.x * NWAVES + wave;
    const int ns = gridDim.x * NWAVES;
    for (int t = wg; t < nTiles; t += ns) {
        const int e = t * 16 + n;
        const bool valid = e < nE;
        const int src = valid ? ei[e] : 0;
        const int dst = valid ? ei[nE + e] : 0;

        // A fragment: row m=n of [x_dst(8), x_src(8), pos_src-pos_dst(2), 0-pad]
        v16h a;
#pragma unroll
        for (int i = 0; i < 16; ++i) a[i] = (_Float16)0.0f;
        if (hi == 0) {
            const float4 x0 = *(const float4*)&x[(size_t)dst * 8];
            const float4 x1 = *(const float4*)&x[(size_t)dst * 8 + 4];
            a[0] = (_Float16)x0.x; a[1] = (_Float16)x0.y;
            a[2] = (_Float16)x0.z; a[3] = (_Float16)x0.w;
            a[4] = (_Float16)x1.x; a[5] = (_Float16)x1.y;
            a[6] = (_Float16)x1.z; a[7] = (_Float16)x1.w;
            const float2 ps = *(const float2*)&pos[(size_t)src * 2];
            const float2 pd = *(const float2*)&pos[(size_t)dst * 2];
            a[8] = (_Float16)(ps.x - pd.x);
            a[9] = (_Float16)(ps.y - pd.y);
        } else {
            const float4 x0 = *(const float4*)&x[(size_t)src * 8];
            const float4 x1 = *(const float4*)&x[(size_t)src * 8 + 4];
            a[0] = (_Float16)x0.x; a[1] = (_Float16)x0.y;
            a[2] = (_Float16)x0.z; a[3] = (_Float16)x0.w;
            a[4] = (_Float16)x1.x; a[5] = (_Float16)x1.y;
            a[6] = (_Float16)x1.z; a[7] = (_Float16)x1.w;
        }

        const float msg = mlp3(a, B0f, B1f, b0, b1, w2s, bias2, hrow, lane);

        if (hi == 0 && valid) {
            atomicAdd(&agg[dst], msg);
            atomicAdd(&cnt[dst], 1.0f);
        }
    }
}

// gamma over nodes + residual
__global__ void __launch_bounds__(256)
gamma_node_kernel(const float* __restrict__ x,
                  const float* __restrict__ agg, const float* __restrict__ cnt,
                  const float* __restrict__ W0, const float* __restrict__ b0,
                  const float* __restrict__ W1, const float* __restrict__ b1,
                  const float* __restrict__ W2, const float* __restrict__ b2,
                  float* __restrict__ out, int nN, int nTiles) {
    __shared__ _Float16 hbuf[NWAVES][16 * HS];
    __shared__ float w2s[64];

    const int tid  = threadIdx.x;
    const int lane = tid & 31;
    const int wave = tid >> 5;
    const int hi   = (lane >> 4) & 1;
    const int n    = lane & 15;

    if (tid < 64) w2s[tid] = W2[tid];
    __syncthreads();

    v16h B0f[4], B1f[2][4];
#pragma unroll
    for (int nt = 0; nt < 4; ++nt) B0f[nt] = load_bfrag(W0, 9, 0, nt * 16, lane);
#pragma unroll
    for (int kk = 0; kk < 2; ++kk)
#pragma unroll
        for (int nt = 0; nt < 4; ++nt)
            B1f[kk][nt] = load_bfrag(W1, 64, kk * 32, nt * 16, lane);
    const float bias2 = b2[0];
    _Float16* hrow = hbuf[wave];

    const int wg = blockIdx.x * NWAVES + wave;
    const int ns = gridDim.x * NWAVES;
    for (int t = wg; t < nTiles; t += ns) {
        const int node = t * 16 + n;
        const bool valid = node < nN;
        const int nd = valid ? node : 0;

        // A fragment: row m=n of [x(8), mean_agg(1), 0-pad]
        v16h a;
#pragma unroll
        for (int i = 0; i < 16; ++i) a[i] = (_Float16)0.0f;
        float xlast = 0.0f;
        if (hi == 0) {
            const float4 x0 = *(const float4*)&x[(size_t)nd * 8];
            const float4 x1 = *(const float4*)&x[(size_t)nd * 8 + 4];
            a[0] = (_Float16)x0.x; a[1] = (_Float16)x0.y;
            a[2] = (_Float16)x0.z; a[3] = (_Float16)x0.w;
            a[4] = (_Float16)x1.x; a[5] = (_Float16)x1.y;
            a[6] = (_Float16)x1.z; a[7] = (_Float16)x1.w;
            xlast = x1.w;
        } else {
            const float am = agg[nd] / fmaxf(cnt[nd], 1.0f);
            a[0] = (_Float16)am;   // k = 8
        }

        const float dudt = mlp3(a, B0f, B1f, b0, b1, w2s, bias2, hrow, lane);

        if (hi == 0 && valid) out[node] = xlast + dudt;
    }
}

// ---------------------------------------------------------------------------
extern "C" void kernel_launch(void* const* d_in, const int* in_sizes, int n_in,
                              void* d_out, int out_size, void* d_ws, size_t ws_size,
                              hipStream_t stream) {
    const float* x      = (const float*)d_in[0];
    const float* pos    = (const float*)d_in[1];
    const int*   ei     = (const int*)d_in[2];
    const float* phi_W0 = (const float*)d_in[3];
    const float* phi_b0 = (const float*)d_in[4];
    const float* phi_W1 = (const float*)d_in[5];
    const float* phi_b1 = (const float*)d_in[6];
    const float* phi_W2 = (const float*)d_in[7];
    const float* phi_b2 = (const float*)d_in[8];
    const float* gam_W0 = (const float*)d_in[9];
    const float* gam_b0 = (const float*)d_in[10];
    const float* gam_W1 = (const float*)d_in[11];
    const float* gam_b1 = (const float*)d_in[12];
    const float* gam_W2 = (const float*)d_in[13];
    const float* gam_b2 = (const float*)d_in[14];

    const int N = in_sizes[0] / 8;   // 100000 nodes
    const int E = in_sizes[2] / 2;   // 1.6M edges

    float* agg = (float*)d_ws;       // [N]
    float* cnt = agg + N;            // [N]

    // zero scratch every launch (determinism; harness does not re-poison)
    const int zn = 2 * N;
    zero_kernel<<<(zn + 255) / 256, 256, 0, stream>>>(agg, zn);

    const int eTiles = (E + 15) >> 4;
    int eBlocks = (eTiles + NWAVES - 1) / NWAVES;
    if (eBlocks > 1024) eBlocks = 1024;
    phi_edge_kernel<<<eBlocks, 256, 0, stream>>>(
        x, pos, ei, phi_W0, phi_b0, phi_W1, phi_b1, phi_W2, phi_b2,
        agg, cnt, E, eTiles);

    const int nTiles = (N + 15) >> 4;
    int nBlocks = (nTiles + NWAVES - 1) / NWAVES;
    if (nBlocks > 1024) nBlocks = 1024;
    gamma_node_kernel<<<nBlocks, 256, 0, stream>>>(
        x, agg, cnt, gam_W0, gam_b0, gam_W1, gam_b1, gam_W2, gam_b2,
        (float*)d_out, N, nTiles);
}
